// HybridGCN_GAT_34076270527038
// MI455X (gfx1250) — matrix operand order repaired and checked by reading, not compiled, not run on hardware
//
#include <hip/hip_runtime.h>
#include <hip/hip_bf16.h>
#include <stdint.h>

#define DD    128
#define NHEAD 2
#define NLAY  3

typedef __attribute__((ext_vector_type(16))) __bf16 v16bf;
typedef __attribute__((ext_vector_type(8)))  __bf16 v8bf;
typedef __attribute__((ext_vector_type(8)))  float  v8f;
typedef unsigned int u32x4 __attribute__((ext_vector_type(4)));
typedef int          i32x8 __attribute__((ext_vector_type(8)));
typedef int          i32x4 __attribute__((ext_vector_type(4)));

// ---------- helpers ----------
__device__ __forceinline__ __bf16 f2bf(float f) {
  union { float f; unsigned u; } c; c.f = f;
  unsigned u = c.u;
  unsigned r = (u + 0x7FFFu + ((u >> 16) & 1u)) >> 16;   // round-to-nearest-even
  union { unsigned short s; __bf16 b; } o; o.s = (unsigned short)r;
  return o.b;
}
__device__ __forceinline__ unsigned fenc(float f) {      // monotone float->uint for atomicMax
  unsigned u = __float_as_uint(f);
  return (u & 0x80000000u) ? ~u : (u | 0x80000000u);
}
__device__ __forceinline__ float fdec(unsigned u) {
  u = (u & 0x80000000u) ? (u & 0x7FFFFFFFu) : ~u;
  return __uint_as_float(u);
}
__device__ __forceinline__ float leaky(float x) { return x > 0.f ? x : 0.2f * x; }
__device__ __forceinline__ void atomAddF(float* p, float v) {
  __hip_atomic_fetch_add(p, v, __ATOMIC_RELAXED, __HIP_MEMORY_SCOPE_AGENT);
}

// ---------- activation f32 -> bf16 (optionally fused ReLU) ----------
__global__ void cvt_bf16_kernel(const float* __restrict__ in, __bf16* __restrict__ out,
                                int n, int doRelu) {
  int i = blockIdx.x * blockDim.x + threadIdx.x;
  if (i >= n) return;
  float v = in[i];
  if (doRelu) v = fmaxf(v, 0.f);
  out[i] = f2bf(v);
}

// ---------- weight packing: W[128, Dout] f32 -> bf16 in WMMA B-operand order ----------
__global__ void pack_w_kernel(const float* __restrict__ W, __bf16* __restrict__ Bp, int Dout) {
  int idx = blockIdx.x * blockDim.x + threadIdx.x;
  if (idx >= DD * Dout) return;
  int j    = idx & 15;
  int lane = (idx >> 4) & 31;
  int kc   = (idx >> 9) & 3;
  int ct   = idx >> 11;
  int k = kc * 32 + (lane >> 4) * 16 + j;
  int n = ct * 16 + (lane & 15);
  Bp[idx] = f2bf(W[(size_t)k * Dout + n]);
}

// ---------- GEMM: C[N, Dout] = A[N,128](bf16) x W(bf16 packed), f32 accum ----------
// block = 256 threads = 8 waves; one block per 32-row tile staged in LDS via one TDM transfer;
// wave wv -> column strip ct, computes TWO 16x16 tiles (rows 0-15 / 16-31) with independent
// accumulator chains, reusing each B fragment twice.
__global__ void gemm_wmma_kernel(const __bf16* __restrict__ Abf, const __bf16* __restrict__ Bp,
                                 float* __restrict__ C, int Dout, int N) {
  __shared__ __bf16 sA[32 * DD];                 // 8 KB: rows rowbase..rowbase+31
  const int wv      = threadIdx.x >> 5;
  const int lane    = threadIdx.x & 31;
  const int rowbase = blockIdx.x << 5;
  const int ct      = blockIdx.y * 8 + wv;       // 16-col tile index
  const int rows    = min(32, N - rowbase);      // 32, or 16 for the tail block

#if __has_builtin(__builtin_amdgcn_tensor_load_to_lds)
  if (wv == 0) {
    // Tensor DMA descriptor (D#): 2D tensor, data_size=2B, tile 128 x rows, stride 128
    unsigned ldsOff = (unsigned)(uintptr_t)(void*)sA;     // generic-low-32 == LDS byte offset
    uint64_t ga = (uint64_t)(uintptr_t)(Abf + (size_t)rowbase * DD);
    u32x4 g0;
    g0[0] = 1u;                                           // count=1 (valid user descriptor)
    g0[1] = ldsOff;                                       // lds_addr
    g0[2] = (unsigned)ga;                                 // global_addr[31:0]
    g0[3] = (unsigned)((ga >> 32) & 0x01FFFFFFu) | (2u << 30);  // global_addr[56:32] | type=2
    i32x8 g1;
    g1[0] = (int)(1u << 16);                  // workgroup_mask=0, data_size=1 (2 bytes)
    g1[1] = (int)(128u << 16);                // tensor_dim0[15:0]=128
    g1[2] = (int)((unsigned)rows << 16);      // tensor_dim0 hi=0 | tensor_dim1[15:0]=rows
    g1[3] = (int)(128u << 16);                // tensor_dim1 hi=0 | tile_dim0=128
    g1[4] = (int)(unsigned)rows;              // tile_dim1=rows | tile_dim2=0
    g1[5] = 128;                              // tensor_dim0_stride[31:0]=128
    g1[6] = 0;                                // stride0 hi | stride1 lo
    g1[7] = 0;                                // stride1 hi
    i32x4 z4 = {0, 0, 0, 0};
#if __clang_major__ >= 23
    i32x8 z8 = {0, 0, 0, 0, 0, 0, 0, 0};
    __builtin_amdgcn_tensor_load_to_lds(g0, g1, z4, z4, z8, 0);
#else
    __builtin_amdgcn_tensor_load_to_lds(g0, g1, z4, z4, 0);
#endif
#if __has_builtin(__builtin_amdgcn_s_wait_tensorcnt)
    __builtin_amdgcn_s_wait_tensorcnt(0);
#endif
  }
  __syncthreads();
#else
  // fallback: cooperative staged load
  for (int i = threadIdx.x; i < rows * DD; i += blockDim.x)
    sA[i] = Abf[(size_t)rowbase * DD + i];
  __syncthreads();
#endif

  const int m  = lane & 15;
  const int hi = lane >> 4;
  const __bf16* arow0 = sA + m * DD;          // rows 0..15 of tile
  const __bf16* arow1 = sA + (16 + m) * DD;   // rows 16..31 of tile
  v8f acc0 = {}, acc1 = {};
#pragma unroll
  for (int kc = 0; kc < 4; ++kc) {
    const int k0 = kc * 32 + hi * 16;
    v16bf b = *(const v16bf*)(Bp + ((((size_t)ct * 4 + kc) * 32 + lane) << 4));
    v8bf lo0 = *(const v8bf*)(arow0 + k0);
    v8bf hb0 = *(const v8bf*)(arow0 + k0 + 8);
    v16bf a0 = __builtin_shufflevector(lo0, hb0, 0, 1, 2, 3, 4, 5, 6, 7,
                                       8, 9, 10, 11, 12, 13, 14, 15);
    acc0 = __builtin_amdgcn_wmma_f32_16x16x32_bf16(false, a0, false, b, (short)0, acc0, false, false);
    v8bf lo1 = *(const v8bf*)(arow1 + k0);
    v8bf hb1 = *(const v8bf*)(arow1 + k0 + 8);
    v16bf a1 = __builtin_shufflevector(lo1, hb1, 0, 1, 2, 3, 4, 5, 6, 7,
                                       8, 9, 10, 11, 12, 13, 14, 15);
    acc1 = __builtin_amdgcn_wmma_f32_16x16x32_bf16(false, a1, false, b, (short)0, acc1, false, false);
  }
  float* cr = C + (size_t)rowbase * Dout + ct * 16 + m;   // N = lane&15
#pragma unroll
  for (int v = 0; v < 8; ++v)
    cr[(size_t)(hi * 8 + v) * Dout] = acc0[v];            // M = (lane>>4)*8 + v
  if (rows == 32) {
#pragma unroll
    for (int v = 0; v < 8; ++v)
      cr[(size_t)(16 + hi * 8 + v) * Dout] = acc1[v];
  }
}

// ---------- degree / normalization ----------
__global__ void deg_init_kernel(float* dinv, int N) {
  int i = blockIdx.x * blockDim.x + threadIdx.x;
  if (i < N) dinv[i] = 1.0f;
}
__global__ void deg_acc_kernel(const int* __restrict__ col, float* dinv, int E) {
  int i = blockIdx.x * blockDim.x + threadIdx.x;
  if (i < E) atomAddF(&dinv[col[i]], 1.0f);
}
__global__ void dinv_kernel(float* dinv, int N) {
  int i = blockIdx.x * blockDim.x + threadIdx.x;
  if (i < N) dinv[i] = rsqrtf(dinv[i]);
}

// ---------- GCN ----------
__global__ void gcn_init_kernel(const float* __restrict__ mm, const float* __restrict__ bias,
                                const float* __restrict__ dinv, float* __restrict__ out, int N) {
  int i = blockIdx.x * blockDim.x + threadIdx.x;
  if (i >= N * DD) return;
  int n = i >> 7, d = i & 127;
  float di = dinv[n];
  out[i] = bias[d] + di * di * mm[i];
}
__global__ void gcn_edge_kernel(const int* __restrict__ row, const int* __restrict__ col,
                                const float* __restrict__ dinv, const float* __restrict__ mm,
                                float* __restrict__ out, int E) {
  int wv = (blockIdx.x * blockDim.x + threadIdx.x) >> 5;   // wave per edge
  int lane = threadIdx.x & 31;
  if (wv >= E) return;
  int r = row[wv], c = col[wv];
  float nrm = dinv[r] * dinv[c];
  float4 v = *(const float4*)(mm + (size_t)r * DD + lane * 4);
  float* o = out + (size_t)c * DD + lane * 4;
  atomAddF(o + 0, nrm * v.x);
  atomAddF(o + 1, nrm * v.y);
  atomAddF(o + 2, nrm * v.z);
  atomAddF(o + 3, nrm * v.w);
}

// ---------- GAT ----------
__global__ void gat_scores_kernel(const float* __restrict__ hp, const float* __restrict__ asrc,
                                  const float* __restrict__ adst, float* __restrict__ es,
                                  float* __restrict__ ed, int N) {
  int wv = (blockIdx.x * blockDim.x + threadIdx.x) >> 5;   // wave per (node, head)
  int lane = threadIdx.x & 31;
  if (wv >= N * NHEAD) return;
  int n = wv >> 1, hh = wv & 1;
  const float* v = hp + (size_t)n * (NHEAD * DD) + hh * DD;
  float s1 = 0.f, s2 = 0.f;
#pragma unroll
  for (int j = lane; j < DD; j += 32) {
    float xv = v[j];
    s1 += xv * asrc[hh * DD + j];
    s2 += xv * adst[hh * DD + j];
  }
#pragma unroll
  for (int off = 16; off; off >>= 1) {
    s1 += __shfl_down(s1, off, 32);
    s2 += __shfl_down(s2, off, 32);
  }
  if (lane == 0) { es[wv] = s1; ed[wv] = s2; }
}
__global__ void gat_maxinit_kernel(const float* __restrict__ es, const float* __restrict__ ed,
                                   unsigned* __restrict__ menc, int N) {
  int i = blockIdx.x * blockDim.x + threadIdx.x;
  if (i < N * NHEAD) menc[i] = fenc(leaky(es[i] + ed[i]));
}
__global__ void gat_edgemax_kernel(const int* __restrict__ row, const int* __restrict__ col,
                                   const float* __restrict__ es, const float* __restrict__ ed,
                                   float* __restrict__ eedge, unsigned* __restrict__ menc, int E) {
  int i = blockIdx.x * blockDim.x + threadIdx.x;
  if (i >= E * NHEAD) return;
  int e = i >> 1, hh = i & 1;
  float ev = leaky(es[row[e] * 2 + hh] + ed[col[e] * 2 + hh]);
  eedge[i] = ev;
  atomicMax(&menc[col[e] * 2 + hh], fenc(ev));
}
__global__ void gat_denominit_kernel(const float* __restrict__ es, const float* __restrict__ ed,
                                     const unsigned* __restrict__ menc, float* __restrict__ denom, int N) {
  int i = blockIdx.x * blockDim.x + threadIdx.x;
  if (i >= N * NHEAD) return;
  denom[i] = expf(leaky(es[i] + ed[i]) - fdec(menc[i]));
}
__global__ void gat_denomacc_kernel(const int* __restrict__ col, const float* __restrict__ eedge,
                                    const unsigned* __restrict__ menc, float* __restrict__ denom, int E) {
  int i = blockIdx.x * blockDim.x + threadIdx.x;
  if (i >= E * NHEAD) return;
  int e = i >> 1, hh = i & 1;
  int c2 = col[e] * 2 + hh;
  atomAddF(&denom[c2], expf(eedge[i] - fdec(menc[c2])));
}
__global__ void gat_agginit_kernel(const float* __restrict__ mm, const float* __restrict__ bias,
                                   const float* __restrict__ es, const float* __restrict__ ed,
                                   const unsigned* __restrict__ menc, const float* __restrict__ denom,
                                   float* __restrict__ out, int N) {
  int i = blockIdx.x * blockDim.x + threadIdx.x;
  if (i >= N * DD) return;
  int n = i >> 7, d = i & 127;
  float acc = bias[d];
#pragma unroll
  for (int hh = 0; hh < NHEAD; ++hh) {
    int i2 = n * 2 + hh;
    float cf = 0.5f * expf(leaky(es[i2] + ed[i2]) - fdec(menc[i2])) / denom[i2];
    acc += cf * mm[(size_t)n * (NHEAD * DD) + hh * DD + d];
  }
  out[i] = acc;
}
__global__ void gat_aggedge_kernel(const int* __restrict__ row, const int* __restrict__ col,
                                   const float* __restrict__ eedge, const unsigned* __restrict__ menc,
                                   const float* __restrict__ denom, const float* __restrict__ mm,
                                   float* __restrict__ out, int E) {
  int wv = (blockIdx.x * blockDim.x + threadIdx.x) >> 5;   // wave per edge
  int lane = threadIdx.x & 31;
  if (wv >= E) return;
  int r = row[wv], c = col[wv];
  float* o = out + (size_t)c * DD + lane * 4;
#pragma unroll
  for (int hh = 0; hh < NHEAD; ++hh) {
    int c2 = c * 2 + hh;
    float cf = 0.5f * expf(eedge[wv * 2 + hh] - fdec(menc[c2])) / denom[c2];
    float4 v = *(const float4*)(mm + (size_t)r * (NHEAD * DD) + hh * DD + lane * 4);
    atomAddF(o + 0, cf * v.x);
    atomAddF(o + 1, cf * v.y);
    atomAddF(o + 2, cf * v.z);
    atomAddF(o + 3, cf * v.w);
  }
}

// ---------- final residual + concat ----------
__global__ void final_kernel(const float* __restrict__ hc, const float* __restrict__ hg,
                             const float* __restrict__ x, float* __restrict__ out, int N) {
  int i = blockIdx.x * blockDim.x + threadIdx.x;
  if (i >= N * DD) return;
  int n = i >> 7, d = i & 127;
  out[(size_t)n * 256 + d]       = hc[i] + x[i];
  out[(size_t)n * 256 + 128 + d] = hg[i] + x[i];
}

extern "C" void kernel_launch(void* const* d_in, const int* in_sizes, int n_in,
                              void* d_out, int out_size, void* d_ws, size_t ws_size,
                              hipStream_t stream) {
  const float* x    = (const float*)d_in[0];
  const int*   ei   = (const int*)d_in[1];
  const float* gcnW = (const float*)d_in[2];
  const float* gcnB = (const float*)d_in[3];
  const float* gatW = (const float*)d_in[4];
  const float* aSrc = (const float*)d_in[5];
  const float* aDst = (const float*)d_in[6];
  const float* gatB = (const float*)d_in[7];
  const int N = in_sizes[0] / DD;
  const int E = in_sizes[1] / 2;
  const int* row = ei;
  const int* col = ei + E;

  // ---- workspace carve-up ----
  float* p = (float*)d_ws;
  float*    dinv  = p;  p += N;
  float*    es    = p;  p += (size_t)N * 2;
  float*    ed    = p;  p += (size_t)N * 2;
  unsigned* menc  = (unsigned*)p; p += (size_t)N * 2;
  float*    denom = p;  p += (size_t)N * 2;
  float*    eedge = p;  p += (size_t)E * 2;
  float*    mm    = p;  p += (size_t)N * 256;   // GEMM output (GCN uses first N*128)
  float*    hc    = p;  p += (size_t)N * DD;    // GCN activations
  float*    hg    = p;  p += (size_t)N * DD;    // GAT activations
  uintptr_t pa = ((uintptr_t)p + 255) & ~(uintptr_t)255;
  __bf16* abf   = (__bf16*)pa;                            // N*128 bf16 activations
  __bf16* wpGcn = abf + (size_t)N * DD;                   // 3*128*128
  __bf16* wpGat = wpGcn + (size_t)NLAY * DD * DD;         // 3*128*256

  const int TB = 256;
  const int gN  = (N * DD + TB - 1) / TB;
  const int gN1 = (N + TB - 1) / TB;
  const int gN2 = (N * 2 + TB - 1) / TB;
  const int gE1 = (E + TB - 1) / TB;
  const int gE2 = (E * 2 + TB - 1) / TB;
  const int gEw = (E + 7) / 8;
  const int gSw = (N * 2 + 7) / 8;
  const int rowTiles32 = (N + 31) / 32;

  // ---- pack weights to bf16 B-operand layout ----
  for (int l = 0; l < NLAY; ++l) {
    pack_w_kernel<<<(DD * DD + TB - 1) / TB, TB, 0, stream>>>(
        gcnW + (size_t)l * DD * DD, wpGcn + (size_t)l * DD * DD, DD);
    pack_w_kernel<<<(DD * 256 + TB - 1) / TB, TB, 0, stream>>>(
        gatW + (size_t)l * DD * 256, wpGat + (size_t)l * DD * 256, 256);
  }

  // ---- degrees / symmetric norm ----
  deg_init_kernel<<<gN1, TB, 0, stream>>>(dinv, N);
  deg_acc_kernel<<<gE1, TB, 0, stream>>>(col, dinv, E);
  dinv_kernel<<<gN1, TB, 0, stream>>>(dinv, N);

  // ---- GCN branch ----
  const float* hin = x;
  for (int l = 0; l < NLAY; ++l) {
    cvt_bf16_kernel<<<gN, TB, 0, stream>>>(hin, abf, N * DD, l > 0);  // fused ReLU for l>0
    gemm_wmma_kernel<<<dim3(rowTiles32, 1), TB, 0, stream>>>(
        abf, wpGcn + (size_t)l * DD * DD, mm, DD, N);
    gcn_init_kernel<<<gN, TB, 0, stream>>>(mm, gcnB + (size_t)l * DD, dinv, hc, N);
    gcn_edge_kernel<<<gEw, TB, 0, stream>>>(row, col, dinv, mm, hc, E);
    hin = hc;
  }

  // ---- GAT branch ----
  const float* gin = x;
  for (int l = 0; l < NLAY; ++l) {
    cvt_bf16_kernel<<<gN, TB, 0, stream>>>(gin, abf, N * DD, l > 0);  // fused ReLU for l>0
    gemm_wmma_kernel<<<dim3(rowTiles32, 2), TB, 0, stream>>>(
        abf, wpGat + (size_t)l * DD * 256, mm, 256, N);
    gat_scores_kernel<<<gSw, TB, 0, stream>>>(mm, aSrc + (size_t)l * NHEAD * DD,
                                              aDst + (size_t)l * NHEAD * DD, es, ed, N);
    gat_maxinit_kernel<<<gN2, TB, 0, stream>>>(es, ed, menc, N);
    gat_edgemax_kernel<<<gE2, TB, 0, stream>>>(row, col, es, ed, eedge, menc, E);
    gat_denominit_kernel<<<gN2, TB, 0, stream>>>(es, ed, menc, denom, N);
    gat_denomacc_kernel<<<gE2, TB, 0, stream>>>(col, eedge, menc, denom, E);
    gat_agginit_kernel<<<gN, TB, 0, stream>>>(mm, gatB + (size_t)l * DD, es, ed, menc, denom, hg, N);
    gat_aggedge_kernel<<<gEw, TB, 0, stream>>>(row, col, eedge, menc, denom, mm, hg, E);
    gin = hg;
  }

  // ---- residual + concat ----
  final_kernel<<<gN, TB, 0, stream>>>(hc, hg, x, (float*)d_out, N);
}
